// AIMNet2InteractionModule_69415261438047
// MI455X (gfx1250) — matrix operand order repaired and checked by reading, not compiled, not run on hardware
//
#include <hip/hip_runtime.h>
#include <math.h>

typedef __attribute__((ext_vector_type(16))) __bf16 v16bf;
typedef __attribute__((ext_vector_type(8)))  __bf16 v8bf;
typedef __attribute__((ext_vector_type(8)))  float  v8f;
typedef __attribute__((ext_vector_type(4)))  int    v4i;

#define WPACK_TOTAL 35328  // 128*160 + 64*128 + 32*64 + 64*32 + 32*64 + 16*32 (shorts)
#define K1 160             // 130 padded to 5 chunks of 32

__device__ __forceinline__ unsigned short f2bf(float f) {
    unsigned int u = __float_as_uint(f);
    unsigned int r = u + 0x7FFFu + ((u >> 16) & 1u);
    return (unsigned short)(r >> 16);
}

__device__ __forceinline__ float gelu_tanh(float x) {
    float x3 = x * x * x;
    return 0.5f * x * (1.0f + tanhf(0.7978845608f * (x + 0.044715f * x3)));
}

// ---------------- Kernel 0: transpose+pad weights to bf16 pack -------------
// Pack layout (shorts): ws1[128][160] ws2[64][128] wa1[32][64] wa2[64][32]
//                       wq1[32][64]  wq2[16][32]   (Wt[n][k] = W[k][n], pads=0)
__global__ __launch_bounds__(256) void prep_weights(
    const float* __restrict__ W1, const float* __restrict__ W2,
    const float* __restrict__ W3, const float* __restrict__ W4,
    const float* __restrict__ W5, const float* __restrict__ W6,
    unsigned short* __restrict__ out)
{
    int stride = gridDim.x * blockDim.x;
    for (int i = blockIdx.x * blockDim.x + threadIdx.x; i < WPACK_TOTAL; i += stride) {
        int r = i;
        float v = 0.0f;
        if (r < 20480)                 { int n = r / 160, k = r % 160; if (k < 130) v = W1[k * 128 + n]; }
        else if ((r -= 20480) <  8192) { int n = r / 128, k = r % 128; v = W2[k * 64 + n]; }
        else if ((r -=  8192) <  2048) { int n = r /  64, k = r %  64; v = W3[k * 32 + n]; }
        else if ((r -=  2048) <  2048) { int n = r /  32, k = r %  32; v = W4[k * 64 + n]; }
        else if ((r -=  2048) <  2048) { int n = r /  64, k = r %  64; v = W5[k * 32 + n]; }
        else { r -= 2048;                int n = r /  32, k = r %  32; if (n < 1) v = W6[k]; }
        out[i] = f2bf(v);
    }
}

// ---------------- Kernel 1: edge gather/multiply/scatter -------------------
// One wave32 per edge; each lane owns two features (float2).
__global__ __launch_bounds__(256) void edge_kernel(
    const float* __restrict__ emb, const float* __restrict__ chg,
    const int* __restrict__ pair, const float* __restrict__ fij,
    const float* __restrict__ rij,
    float* __restrict__ rad_e, float* __restrict__ vec_e,
    float* __restrict__ rad_c, float* __restrict__ vec_c, int E)
{
    int gwave = blockIdx.x * 8 + (threadIdx.x >> 5);
    int lane  = threadIdx.x & 31;
    if (gwave >= E) return;
    int j = pair[(size_t)E + gwave];                 // idx_j = pair_indices[1][e]
    float2 f2 = *(const float2*)(fij + (size_t)gwave * 64 + lane * 2);
    float2 e2 = *(const float2*)(emb + (size_t)j * 64 + lane * 2);
    float qj  = chg[j];
    float p0 = f2.x * e2.x, p1 = f2.y * e2.y;
    size_t rb = (size_t)j * 64 + lane * 2;
    unsafeAtomicAdd(&rad_e[rb],     p0);
    unsafeAtomicAdd(&rad_e[rb + 1], p1);
    unsafeAtomicAdd(&rad_c[rb],     f2.x * qj);
    unsafeAtomicAdd(&rad_c[rb + 1], f2.y * qj);
    float s = p0 + p1, sf = f2.x + f2.y;             // row sums of proto_e / f_ij
#pragma unroll
    for (int m = 16; m > 0; m >>= 1) { s += __shfl_xor(s, m, 32); sf += __shfl_xor(sf, m, 32); }
    if (lane < 3) {
        float r = rij[(size_t)gwave * 3 + lane];
        unsafeAtomicAdd(&vec_e[(size_t)j * 3 + lane], r * s);
        unsafeAtomicAdd(&vec_c[(size_t)j * 3 + lane], r * sf * qj);
    }
}

// ---------------- Kernel 2: WMMA MLP -------------------------------------
// A fragment (16x32 bf16), contiguous layout (matches load_b):
//   lane<16 : row=lane,    elem e -> k = k0 + e
//   lane>=16: row=lane-16, elem e -> k = k0 + 16 + e
__device__ __forceinline__ v16bf load_a(const unsigned short* buf, int ldk, int k0, int lane) {
    const int half = lane >> 4, row = lane & 15;
    const unsigned short* p = buf + row * ldk + k0 + 16 * half;
    v8bf lo = *(const v8bf*)p;
    v8bf hi = *(const v8bf*)(p + 8);
    return __builtin_shufflevector(lo, hi, 0, 1, 2, 3, 4, 5, 6, 7, 8, 9, 10, 11, 12, 13, 14, 15);
}

// B fragment (32x16 bf16): lane holds 16 contiguous K for column (lane&15),
// K base k0 + 16*(lane>>4). Weights stored transposed: wt[n][K].
__device__ __forceinline__ v16bf load_b(const unsigned short* wt, int K, int k0, int n0, int lane) {
    const int half = lane >> 4, col = lane & 15;
    const unsigned short* p = wt + (size_t)(n0 + col) * K + k0 + 16 * half;
    return *(const v16bf*)p;
}

__global__ __launch_bounds__(128) void mlp_kernel(
    const float* __restrict__ rad_e, const float* __restrict__ vec_e,
    const float* __restrict__ rad_c, const float* __restrict__ vec_c,
    const unsigned short* __restrict__ wpack,
    const float* __restrict__ bs1, const float* __restrict__ bs2,
    const float* __restrict__ ba1, const float* __restrict__ ba2,
    const float* __restrict__ bq1, const float* __restrict__ bq2,
    float* __restrict__ out, int nAtoms)
{
    __shared__ __align__(32) unsigned short smem[WPACK_TOTAL + 4 * 16 * K1];
    const int tid = threadIdx.x, lane = tid & 31, w = tid >> 5;

    // ---- stage weight pack into LDS (async global->LDS when available) ----
#if __has_builtin(__builtin_amdgcn_global_load_async_to_lds_b128)
    for (int i = tid * 8; i < WPACK_TOTAL; i += blockDim.x * 8) {
        __builtin_amdgcn_global_load_async_to_lds_b128(
            (__attribute__((address_space(1))) v4i*)(wpack + i),
            (__attribute__((address_space(3))) v4i*)(smem + i), 0, 0);
    }
#if __has_builtin(__builtin_amdgcn_s_wait_asynccnt)
    __builtin_amdgcn_s_wait_asynccnt(0);
#else
    asm volatile("s_wait_asynccnt 0" ::: "memory");
#endif
#else
    for (int i = tid * 8; i < WPACK_TOTAL; i += blockDim.x * 8)
        *(uint4*)(smem + i) = *(const uint4*)(wpack + i);
#endif

    unsigned short* ws1 = smem;
    unsigned short* ws2 = smem + 20480;
    unsigned short* wa1 = smem + 28672;
    unsigned short* wa2 = smem + 30720;
    unsigned short* wq1 = smem + 32768;
    unsigned short* wq2 = smem + 34816;
    unsigned short* msg = smem + WPACK_TOTAL + w * (16 * K1);

    // ---- build msg tile [16][160] bf16 for this wave's 16 atoms ----
    const int base = (blockIdx.x * 4 + w) * 16;
    for (int i = lane; i < 16 * K1; i += 32) {
        int r = i / K1, c = i - r * K1;
        int atom = base + r;
        float v = 0.0f;
        if (atom < nAtoms) {
            if (c < 64)        v = rad_e[(size_t)atom * 64 + c];
            else if (c == 64)  { const float* ve = vec_e + (size_t)atom * 3;
                                 v = sqrtf(ve[0]*ve[0] + ve[1]*ve[1] + ve[2]*ve[2]); }
            else if (c < 129)  v = rad_c[(size_t)atom * 64 + (c - 65)];
            else if (c == 129) { const float* vc = vec_c + (size_t)atom * 3;
                                 v = sqrtf(vc[0]*vc[0] + vc[1]*vc[1] + vc[2]*vc[2]); }
        }
        msg[i] = f2bf(v);
    }
    __syncthreads();

    const int half = lane >> 4, col = lane & 15;

    // ---- layer 1: [16,160] x [160,128] -> gelu -> h1 [16,128] ----
    v16bf A1[5];
#pragma unroll
    for (int k = 0; k < 5; k++) A1[k] = load_a(msg, K1, k * 32, lane);
    unsigned short* h1 = msg;   // A1 cached in regs; safe to overwrite
#pragma unroll
    for (int n = 0; n < 8; n++) {
        float bv = bs1[n * 16 + col];
        v8f acc; for (int e = 0; e < 8; e++) acc[e] = bv;
#pragma unroll
        for (int k = 0; k < 5; k++)
            acc = __builtin_amdgcn_wmma_f32_16x16x32_bf16(
                false, A1[k], false, load_b(ws1, K1, k * 32, n * 16, lane),
                (short)0, acc, false, false);
        for (int e = 0; e < 8; e++)
            h1[(e + 8 * half) * 128 + n * 16 + col] = f2bf(gelu_tanh(acc[e]));
    }

    // ---- layer 2: [16,128] x [128,64] -> gelu -> h2 [16,64] ----
    v16bf A2[4];
#pragma unroll
    for (int k = 0; k < 4; k++) A2[k] = load_a(h1, 128, k * 32, lane);
    unsigned short* h2 = msg;
#pragma unroll
    for (int n = 0; n < 4; n++) {
        float bv = bs2[n * 16 + col];
        v8f acc; for (int e = 0; e < 8; e++) acc[e] = bv;
#pragma unroll
        for (int k = 0; k < 4; k++)
            acc = __builtin_amdgcn_wmma_f32_16x16x32_bf16(
                false, A2[k], false, load_b(ws2, 128, k * 32, n * 16, lane),
                (short)0, acc, false, false);
        for (int e = 0; e < 8; e++)
            h2[(e + 8 * half) * 64 + n * 16 + col] = f2bf(gelu_tanh(acc[e]));
    }

    v16bf Ah[2];
    Ah[0] = load_a(h2, 64, 0,  lane);
    Ah[1] = load_a(h2, 64, 32, lane);

    // ---- a-head: gelu([16,64]x[64,32]) x [32,64] + b -> delta_a ----
    unsigned short* ta = msg;
#pragma unroll
    for (int n = 0; n < 2; n++) {
        float bv = ba1[n * 16 + col];
        v8f acc; for (int e = 0; e < 8; e++) acc[e] = bv;
#pragma unroll
        for (int k = 0; k < 2; k++)
            acc = __builtin_amdgcn_wmma_f32_16x16x32_bf16(
                false, Ah[k], false, load_b(wa1, 64, k * 32, n * 16, lane),
                (short)0, acc, false, false);
        for (int e = 0; e < 8; e++)
            ta[(e + 8 * half) * 32 + n * 16 + col] = f2bf(gelu_tanh(acc[e]));
    }
    v16bf Aa = load_a(ta, 32, 0, lane);
#pragma unroll
    for (int n = 0; n < 4; n++) {
        float bv = ba2[n * 16 + col];
        v8f acc; for (int e = 0; e < 8; e++) acc[e] = bv;
        acc = __builtin_amdgcn_wmma_f32_16x16x32_bf16(
            false, Aa, false, load_b(wa2, 32, 0, n * 16, lane),
            (short)0, acc, false, false);
        for (int e = 0; e < 8; e++) {
            int atom = base + e + 8 * half;
            if (atom < nAtoms) out[(size_t)atom * 64 + n * 16 + col] = acc[e];
        }
    }

    // ---- q-head: gelu([16,64]x[64,32]) x [32,1(->16 pad)] + b -> delta_q ----
    unsigned short* tq = msg;
#pragma unroll
    for (int n = 0; n < 2; n++) {
        float bv = bq1[n * 16 + col];
        v8f acc; for (int e = 0; e < 8; e++) acc[e] = bv;
#pragma unroll
        for (int k = 0; k < 2; k++)
            acc = __builtin_amdgcn_wmma_f32_16x16x32_bf16(
                false, Ah[k], false, load_b(wq1, 64, k * 32, n * 16, lane),
                (short)0, acc, false, false);
        for (int e = 0; e < 8; e++)
            tq[(e + 8 * half) * 32 + n * 16 + col] = f2bf(gelu_tanh(acc[e]));
    }
    v16bf Aq = load_a(tq, 32, 0, lane);
    {
        float bv = (col == 0) ? bq2[0] : 0.0f;
        v8f acc; for (int e = 0; e < 8; e++) acc[e] = bv;
        acc = __builtin_amdgcn_wmma_f32_16x16x32_bf16(
            false, Aq, false, load_b(wq2, 32, 0, 0, lane),
            (short)0, acc, false, false);
        if (col == 0)
            for (int e = 0; e < 8; e++) {
                int atom = base + e + 8 * half;
                if (atom < nAtoms) out[(size_t)nAtoms * 64 + atom] = acc[e];
            }
    }
}

// ---------------- host-side launch ----------------------------------------
extern "C" void kernel_launch(void* const* d_in, const int* in_sizes, int n_in,
                              void* d_out, int out_size, void* d_ws, size_t ws_size,
                              hipStream_t stream) {
    const float* emb = (const float*)d_in[0];
    const float* chg = (const float*)d_in[1];
    const int*   pr  = (const int*)  d_in[2];
    const float* fij = (const float*)d_in[3];
    const float* rij = (const float*)d_in[4];
    const float* Ws1 = (const float*)d_in[5];  const float* bs1 = (const float*)d_in[6];
    const float* Ws2 = (const float*)d_in[7];  const float* bs2 = (const float*)d_in[8];
    const float* Wa1 = (const float*)d_in[9];  const float* ba1 = (const float*)d_in[10];
    const float* Wa2 = (const float*)d_in[11]; const float* ba2 = (const float*)d_in[12];
    const float* Wq1 = (const float*)d_in[13]; const float* bq1 = (const float*)d_in[14];
    const float* Wq2 = (const float*)d_in[15]; const float* bq2 = (const float*)d_in[16];

    const int N = in_sizes[1];          // partial_charges is [N,1]
    const int E = in_sizes[4] / 3;      // r_ij is [E,3]

    float* acc = (float*)d_ws;
    float* rad_e = acc;
    float* vec_e = acc + (size_t)N * 64;
    float* rad_c = acc + (size_t)N * 67;
    float* vec_c = acc + (size_t)N * 131;
    size_t accBytes = (size_t)N * 134 * sizeof(float);
    size_t offW = (accBytes + 255) & ~(size_t)255;
    unsigned short* wpack = (unsigned short*)((char*)d_ws + offW);

    (void)hipMemsetAsync(d_ws, 0, accBytes, stream);
    prep_weights<<<32, 256, 0, stream>>>(Ws1, Ws2, Wa1, Wa2, Wq1, Wq2, wpack);
    edge_kernel<<<(E + 7) / 8, 256, 0, stream>>>(emb, chg, pr, fij, rij,
                                                 rad_e, vec_e, rad_c, vec_c, E);
    mlp_kernel<<<(N + 63) / 64, 128, 0, stream>>>(rad_e, vec_e, rad_c, vec_c, wpack,
                                                  bs1, bs2, ba1, ba2, bq1, bq2,
                                                  (float*)d_out, N);
}
